// SumLSTM_37477884625383
// MI455X (gfx1250) — matrix operand order repaired and checked by reading, not compiled, not run on hardware
//
#include <hip/hip_runtime.h>

// SumLSTM on MI455X (gfx1250, wave32, WMMA) — transpose-free recurrence.
//
// One wave owns a 16-row batch tile for all 512 steps. Per step:
//   gates^T[64x16] = W_aug[64x32] x [h | x | 1]_aug[32x16]
// as 4x v_wmma_f32_16x16x32_f16 (tile n = gate n: i,f,g,o since H==16).
//
// K-slot assignment (shared by A and B):
//   K0..7  = hidden 0..7      K8 = x_t     K9 = 1.0     K10..15 = 0
//   K16..23= hidden 8..15     K24..31 = 0
// With this map, the WMMA D layout (lane l: gate-hidden v+8*(l>>4),
// batch l&15) IS the B layout of the next step:
//   B lane j<16 : halves 0..7 = K0..7   = h[b=j][0..7]   (this lane's own D)
//   B lane j>=16: halves 0..7 = K16..23 = h[b=j][8..15]  (this lane's own D)
// -> the h feedback is 4 v_cvt_pk_f16_f32 per step; no LDS, no shuffles.
//
// A (weights) is built once:
//   lane<16 (M=row): halves 0..7 = W_hh[g][0..7] (K0..7), 8,9 = W_ih[g], b_ih[g]+b_hh[g] (K8,K9); rest 0
//   lane>=16       : halves 0..7  = W_hh[g][8..15] (K16..23); rest 0
//
// c stays f32 in registers (same layout as D); only WMMA operands are f16
// with f32 accumulation. Activations use v_exp_f32/v_rcp_f32 (TRANS pipe,
// co-executes with the XDL WMMAs).

typedef __attribute__((ext_vector_type(16))) _Float16 v16h;
typedef __attribute__((ext_vector_type(8)))  float    v8f;
typedef __attribute__((ext_vector_type(4)))  float    f4;

#define TSTEPS 512

__device__ __forceinline__ float fast_sigmoid(float x) {
    // 1 / (1 + 2^(-x*log2e))
    float e = __builtin_amdgcn_exp2f(x * -1.4426950408889634f);
    return __builtin_amdgcn_rcpf(1.0f + e);
}
__device__ __forceinline__ float fast_tanh(float x) {
    // 2/(1 + 2^(-2x*log2e)) - 1
    float e = __builtin_amdgcn_exp2f(x * -2.8853900817779268f);
    return __builtin_fmaf(2.0f, __builtin_amdgcn_rcpf(1.0f + e), -1.0f);
}

__global__ __launch_bounds__(32)
void lstm16_wmma_kernel(const float* __restrict__ x,
                        const float* __restrict__ W_ih,
                        const float* __restrict__ W_hh,
                        const float* __restrict__ b_ih,
                        const float* __restrict__ b_hh,
                        const float* __restrict__ W_lin,
                        float* __restrict__ out)
{
    const int lane = threadIdx.x;       // 0..31 (wave32)
    const int col  = lane & 15;
    const int hi   = lane >> 4;         // 0: lanes 0-15, 1: lanes 16-31
    const int tile = blockIdx.x;        // batch tile (16 rows)

    // ---------------- A operands (weights), built once ----------------------
    v16h Am[4];
#pragma unroll
    for (int n = 0; n < 4; ++n) {
        const int g = n * 16 + col;              // gate row (M = col)
        v16h a = {};
        {
            const float* row = W_hh + g * 16 + 8 * hi;    // 8 floats: hidden 8*hi..8*hi+7
#pragma unroll
            for (int e = 0; e < 8; ++e) a[e] = (_Float16)row[e];
        }
        if (hi == 0) {
            a[8] = (_Float16)W_ih[g];                    // K16 : x coeff
            a[9] = (_Float16)(b_ih[g] + b_hh[g]);        // K17 : bias
        }
        Am[n] = a;
    }

    // ---------------- loop-carried state ------------------------------------
    v8f  cst  = {};              // cell state f32, D layout
    v16h bmat = {};              // B operand: halves 0..7 = h (init 0)
    if (hi == 0) bmat[9] = (_Float16)1.0f;   // K9 = 1.0, constant all steps

    // every lane reads the x row matching its batch column (lanes 16-31
    // duplicate lanes 0-15; their x slot is unused) -> no OOB
    const float* xrow = x + (size_t)(tile * 16 + col) * TSTEPS;

    for (int t0 = 0; t0 < TSTEPS; t0 += 4) {
        f4 xq = *(const f4*)(xrow + t0);                  // global_load_b128
        if ((t0 & 63) == 0 && (t0 + 64) < TSTEPS)
            __builtin_prefetch(xrow + t0 + 64, 0, 0);     // global_prefetch

#pragma unroll
        for (int s = 0; s < 4; ++s) {
            const float xv = (s == 0) ? xq.x : (s == 1) ? xq.y
                           : (s == 2) ? xq.z : xq.w;

            if (hi == 0) bmat[8] = (_Float16)xv;          // K8 = x_t

            // ---- gates^T = A x B (+0): one WMMA per gate -----------------
            v8f z = {};
            v8f di = __builtin_amdgcn_wmma_f32_16x16x32_f16(false, Am[0], false, bmat, (short)0, z, false, false);
            v8f df = __builtin_amdgcn_wmma_f32_16x16x32_f16(false, Am[1], false, bmat, (short)0, z, false, false);
            v8f dg = __builtin_amdgcn_wmma_f32_16x16x32_f16(false, Am[2], false, bmat, (short)0, z, false, false);
            v8f dz = __builtin_amdgcn_wmma_f32_16x16x32_f16(false, Am[3], false, bmat, (short)0, z, false, false);

            // ---- elementwise LSTM cell update; h feeds straight back ------
#pragma unroll
            for (int v = 0; v < 8; ++v) {
                float iv = fast_sigmoid(di[v]);
                float fv = fast_sigmoid(df[v]);
                float gv = fast_tanh(dg[v]);
                float ov = fast_sigmoid(dz[v]);
                float cc = __builtin_fmaf(fv, cst[v], iv * gv);
                cst[v] = cc;
                float hv = ov * fast_tanh(cc);
                bmat[v] = (_Float16)hv;      // D layout == next-step B layout
            }
        }
    }

    // ---- out[b] = h_last[b,:] . W_lin --------------------------------------
    // lane l holds h[b=l&15][k=v+8*hi]; partial-dot per lane, then combine
    // lane pairs (j, j+16) with one ds_swizzle SWAPX16.
    float part = 0.0f;
#pragma unroll
    for (int v = 0; v < 8; ++v)
        part = __builtin_fmaf((float)bmat[v], W_lin[v + 8 * hi], part);

    int other_i = __builtin_amdgcn_ds_swizzle(__float_as_int(part), 0x401f); // SWAPX16
    float other = __int_as_float(other_i);
    if (hi == 0)
        out[tile * 16 + col] = part + other;
}

extern "C" void kernel_launch(void* const* d_in, const int* in_sizes, int n_in,
                              void* d_out, int out_size, void* d_ws, size_t ws_size,
                              hipStream_t stream) {
    const float* x     = (const float*)d_in[0];  // [B, 512, 1]
    const float* W_ih  = (const float*)d_in[1];  // [64, 1]
    const float* W_hh  = (const float*)d_in[2];  // [64, 16]
    const float* b_ih  = (const float*)d_in[3];  // [64]
    const float* b_hh  = (const float*)d_in[4];  // [64]
    const float* W_lin = (const float*)d_in[5];  // [1, 16]
    float* out = (float*)d_out;                  // [B, 1]

    const int B = in_sizes[0] / TSTEPS;          // x flat = B*T*1
    const int tiles = B / 16;                    // 256 single-wave workgroups
    lstm16_wmma_kernel<<<tiles, 32, 0, stream>>>(x, W_ih, W_hh, b_ih, b_hh,
                                                 W_lin, out);
}
